// TransformerBlock_63677185130746
// MI455X (gfx1250) — matrix operand order repaired and checked by reading, not compiled, not run on hardware
//
#include <hip/hip_runtime.h>
#include <stdint.h>

// ---------------- problem constants ----------------
constexpr int Dm = 1024;      // d_model
constexpr int Dh = 4096;      // d_hidden
constexpr int Bt = 4;         // batch
constexpr int S  = 2048;      // seq
constexpr float EPS        = 1e-5f;
constexpr float SLOPE      = 0.01f;
constexpr float INV_SQRT_D = 0.03125f;   // 1/sqrt(1024)

// ---------------- GEMM tiling ----------------
constexpr int BM = 128, BN = 128, BK = 32;
constexpr int LDSK = 40;      // 32 + 8 bf16 pad (80B row, 16B-multiple)

typedef __attribute__((ext_vector_type(16))) __bf16 v16bf;
typedef __attribute__((ext_vector_type(8)))  float  v8f;

#define DEV static __device__ __forceinline__

DEV __bf16 f2bf(float f) {
    uint32_t u = __builtin_bit_cast(uint32_t, f);
    u += 0x7FFFu + ((u >> 16) & 1u);          // round-to-nearest-even
    uint16_t h = (uint16_t)(u >> 16);
    return __builtin_bit_cast(__bf16, h);
}

union FragU { uint4 q[2]; v16bf v; };

// A-matrix 16x32 bf16 fragment (ISA 7.12.2): lane%16 = row M,
// lane/16 selects K chunks {0..7,16..23} vs {8..15,24..31}.
DEV v16bf load_a_frag(const __bf16* rowp, int half) {
    FragU f;
    f.q[0] = *(const uint4*)(rowp + half * 8);
    f.q[1] = *(const uint4*)(rowp + 16 + half * 8);
    return f.v;
}
// B-matrix 32x16 bf16 fragment: lane%16 = col N, lane/16 selects K chunks
// {0..7,16..23} vs {8..15,24..31} (tile stored [N][K]), matching A.
DEV v16bf load_b_frag(const __bf16* rowp, int half) {
    FragU f;
    f.q[0] = *(const uint4*)(rowp + half * 8);
    f.q[1] = *(const uint4*)(rowp + 16 + half * 8);
    return f.v;
}

// ---- CDNA5 async global->LDS copy (ASYNCcnt path, ISA §15.18 op 98) ----
// 16 bytes per lane, GV addressing: vdst = LDS byte offset (low 32 bits of
// the generic LDS pointer per the aperture rules), vaddr = 64-bit address.
DEV void async_copy_b128(const void* lds, const void* gptr) {
    uint32_t ldsOff = (uint32_t)(uintptr_t)lds;
    asm volatile("global_load_async_to_lds_b128 %0, %1, off"
                 :: "v"(ldsOff), "v"(gptr)
                 : "memory");
}
DEV void wait_async0() {
    asm volatile("s_wait_asynccnt 0x0" ::: "memory");
}

// epilogue modes
constexpr int OUT_BF16      = 0;  // bf16 C[M][N] (+bias +leaky optional)
constexpr int OUT_BF16_T    = 1;  // bf16 transposed per batch: V^T[b][n][t]
constexpr int OUT_SCORES    = 2;  // f32 C = acc/sqrt(D), per-batch
constexpr int OUT_F32_RESID = 3;  // f32 C = resid + (act(acc+bias))

// ============================================================
// C[M,N] = A[M,K] (bf16, row-major) @ Bw[N,K]^T (bf16, row-major)
// 256 threads = 8 waves; block tile 128x128; wave tile 32x64;
// 8 x v_wmma_f32_16x16x32_bf16 per 32-wide k-step.
// LDS double buffered via GLOBAL_LOAD_ASYNC_TO_LDS_B128: next tile's
// async loads overlap current tile's WMMAs; one barrier per k-step.
// ============================================================
template <int OUTM, bool BIAS, bool ACT, bool CSKIP, bool CKLIM>
__global__ __launch_bounds__(256)
void gemm_bf16_wmma(const __bf16* __restrict__ A,
                    const __bf16* __restrict__ Bw,
                    const float*  __restrict__ bias,
                    const float*  __restrict__ resid,
                    void*         __restrict__ Cout,
                    int M, int N, int K,
                    long aBat, long bBat, long cBat, long rBat) {
    const int m0 = blockIdx.y * BM;
    const int n0 = blockIdx.x * BN;
    if (CSKIP && n0 >= m0 + BM) return;          // fully-masked causal block
    const int z = blockIdx.z;
    A  += (long)z * aBat;
    Bw += (long)z * bBat;

    __shared__ __align__(16) __bf16 aSh[2][BM][LDSK];
    __shared__ __align__(16) __bf16 bSh[2][BN][LDSK];

    const int tid  = threadIdx.x;
    const int lane = tid & 31, wave = tid >> 5;
    const int wm = wave >> 1, wn = wave & 1;     // 4x2 wave grid
    const int half = lane >> 4, l16 = lane & 15;

    v8f acc[2][4];
#pragma unroll
    for (int i = 0; i < 2; ++i)
#pragma unroll
        for (int j = 0; j < 4; ++j)
#pragma unroll
            for (int r = 0; r < 8; ++r) acc[i][j][r] = 0.0f;

    // cooperative tile loaders: 256 thr x 16 bf16 = 128x32 tile
    const int lr = tid >> 1;
    const int lc = (tid & 1) << 4;
    const __bf16* aG = A  + (long)(m0 + lr) * K + lc;
    const __bf16* bG = Bw + (long)(n0 + lr) * K + lc;

    int kEnd = K;
    if (CKLIM) kEnd = (K < m0 + BM) ? K : (m0 + BM);  // causal K clamp
    const int nK = kEnd >> 5;

    // k-tile 0 -> buffer 0 (async direct-to-LDS)
    async_copy_b128(&aSh[0][lr][lc],     aG);
    async_copy_b128(&aSh[0][lr][lc + 8], aG + 8);
    async_copy_b128(&bSh[0][lr][lc],     bG);
    async_copy_b128(&bSh[0][lr][lc + 8], bG + 8);
    wait_async0();
    __syncthreads();

    for (int kt = 0; kt < nK; ++kt) {
        const int cur = kt & 1, nxt = cur ^ 1;
        if (kt + 1 < nK) {                       // overlap next tile with WMMAs
            const __bf16* ag = aG + (kt + 1) * BK;
            const __bf16* bg = bG + (kt + 1) * BK;
            async_copy_b128(&aSh[nxt][lr][lc],     ag);
            async_copy_b128(&aSh[nxt][lr][lc + 8], ag + 8);
            async_copy_b128(&bSh[nxt][lr][lc],     bg);
            async_copy_b128(&bSh[nxt][lr][lc + 8], bg + 8);
        }

        v16bf af[2], bfr[4];
#pragma unroll
        for (int mt = 0; mt < 2; ++mt)
            af[mt] = load_a_frag(&aSh[cur][wm * 32 + mt * 16 + l16][0], half);
#pragma unroll
        for (int nt = 0; nt < 4; ++nt)
            bfr[nt] = load_b_frag(&bSh[cur][wn * 64 + nt * 16 + l16][0], half);

#pragma unroll
        for (int mt = 0; mt < 2; ++mt)
#pragma unroll
            for (int nt = 0; nt < 4; ++nt)
                acc[mt][nt] = __builtin_amdgcn_wmma_f32_16x16x32_bf16(
                    false, af[mt], false, bfr[nt],
                    (short)0, acc[mt][nt], false, false);

        wait_async0();                            // next tile landed in LDS
        __syncthreads();                          // visible to all waves
    }

    // ---------------- epilogue (C layout per ISA 7.12.2) ----------------
    float*  Cf = (float*)Cout;
    __bf16* Cb = (__bf16*)Cout;
    const float* Rf = resid;
    if (OUTM == OUT_SCORES || OUTM == OUT_F32_RESID) Cf += (long)z * cBat;
    if (OUTM == OUT_F32_RESID) Rf = resid + (long)z * rBat;

#pragma unroll
    for (int mt = 0; mt < 2; ++mt) {
#pragma unroll
        for (int nt = 0; nt < 4; ++nt) {
            const int row0 = m0 + wm * 32 + mt * 16 + half * 8;
            const int col  = n0 + wn * 64 + nt * 16 + l16;
            float bv = 0.0f;
            if (BIAS) bv = bias[col];
#pragma unroll
            for (int r = 0; r < 8; ++r) {
                const int grow = row0 + r;
                float v = acc[mt][nt][r];
                if (OUTM == OUT_SCORES) {
                    Cf[(long)grow * N + col] = v * INV_SQRT_D;
                } else if (OUTM == OUT_BF16) {
                    v += bv;
                    if (ACT) v = (v >= 0.0f) ? v : v * SLOPE;
                    Cb[(long)grow * N + col] = f2bf(v);
                } else if (OUTM == OUT_BF16_T) {
                    const int bb = grow >> 11;        // / S
                    const int tt = grow & (S - 1);
                    Cb[((long)bb * Dm + col) * S + tt] = f2bf(v);
                } else {                              // OUT_F32_RESID
                    v += bv;
                    if (ACT) v = (v >= 0.0f) ? v : v * SLOPE;
                    const long idx = (long)grow * N + col;
                    Cf[idx] = Rf[idx] + v;
                }
            }
        }
    }
}

// ============================================================
// LayerNorm of one 1024-wide row -> bf16
// ============================================================
__global__ __launch_bounds__(256)
void ln_bf16_kernel(const float* __restrict__ x, const float* __restrict__ g,
                    const float* __restrict__ b, __bf16* __restrict__ out) {
    const long row = blockIdx.x;
    const int  t   = threadIdx.x;
    float4 v = ((const float4*)(x + row * Dm))[t];
    float s = v.x + v.y + v.z + v.w;
    float q = v.x * v.x + v.y * v.y + v.z * v.z + v.w * v.w;
    __shared__ float ss[256], sq[256];
    ss[t] = s; sq[t] = q;
    __syncthreads();
    for (int off = 128; off; off >>= 1) {
        if (t < off) { ss[t] += ss[t + off]; sq[t] += sq[t + off]; }
        __syncthreads();
    }
    const float mean = ss[0] * (1.0f / Dm);
    const float var  = sq[0] * (1.0f / Dm) - mean * mean;
    const float rstd = rsqrtf(var + EPS);
    float4 gv = ((const float4*)g)[t];
    float4 bv = ((const float4*)b)[t];
    __bf16* o = out + row * Dm + t * 4;
    o[0] = f2bf((v.x - mean) * rstd * gv.x + bv.x);
    o[1] = f2bf((v.y - mean) * rstd * gv.y + bv.y);
    o[2] = f2bf((v.z - mean) * rstd * gv.z + bv.z);
    o[3] = f2bf((v.w - mean) * rstd * gv.w + bv.w);
}

// ============================================================
// Causal row softmax over 2048 f32 scores -> bf16 probs.
// Index-masked: never reads the skipped upper-triangular blocks.
// ============================================================
__global__ __launch_bounds__(256)
void softmax_bf16_kernel(const float* __restrict__ scores,
                         __bf16* __restrict__ probs) {
    const long row = blockIdx.x;                 // b*S + s
    const int  s   = (int)(row & (S - 1));
    const float* sr = scores + row * (long)S;
    __bf16*      pr = probs  + row * (long)S;
    const int t = threadIdx.x;
    float v[8];
    float mx = -1e30f;
#pragma unroll
    for (int i = 0; i < 8; ++i) {
        const int j = t + i * 256;
        v[i] = (j <= s) ? sr[j] : -1e30f;
        mx = fmaxf(mx, v[i]);
    }
    __shared__ float red[256];
    red[t] = mx; __syncthreads();
    for (int off = 128; off; off >>= 1) {
        if (t < off) red[t] = fmaxf(red[t], red[t + off]);
        __syncthreads();
    }
    mx = red[0]; __syncthreads();
    float sum = 0.0f;
#pragma unroll
    for (int i = 0; i < 8; ++i) { v[i] = __expf(v[i] - mx); sum += v[i]; }
    red[t] = sum; __syncthreads();
    for (int off = 128; off; off >>= 1) {
        if (t < off) red[t] += red[t + off];
        __syncthreads();
    }
    const float inv = 1.0f / red[0];
#pragma unroll
    for (int i = 0; i < 8; ++i) pr[t + i * 256] = f2bf(v[i] * inv);
}

// ---------------- f32 -> bf16 converters ----------------
__global__ void cvt_bf16_kernel(const float* __restrict__ in,
                                __bf16* __restrict__ out, long n) {
    long i = (long)blockIdx.x * blockDim.x + threadIdx.x;
    const long stride = (long)gridDim.x * blockDim.x;
    for (; i < n; i += stride) out[i] = f2bf(in[i]);
}
// out[c*rows + r] = bf16(in[r*cols + c])  (for qk / ov -> [N][K] layout)
__global__ void cvtT_bf16_kernel(const float* __restrict__ in,
                                 __bf16* __restrict__ out,
                                 int rows, int cols) {
    __shared__ float tile[32][33];
    const int r0 = blockIdx.y * 32, c0 = blockIdx.x * 32;
    const int tx = threadIdx.x, ty = threadIdx.y;     // (32, 8)
    for (int i = 0; i < 32; i += 8)
        tile[ty + i][tx] = in[(long)(r0 + ty + i) * cols + c0 + tx];
    __syncthreads();
    for (int i = 0; i < 32; i += 8)
        out[(long)(c0 + ty + i) * rows + r0 + tx] = f2bf(tile[tx][ty + i]);
}

// ============================================================
extern "C" void kernel_launch(void* const* d_in, const int* in_sizes, int n_in,
                              void* d_out, int out_size, void* d_ws, size_t ws_size,
                              hipStream_t stream) {
    const float* x    = (const float*)d_in[0];
    const float* qk   = (const float*)d_in[1];
    const float* ov   = (const float*)d_in[2];
    const float* ln1g = (const float*)d_in[3];
    const float* ln1b = (const float*)d_in[4];
    const float* ln2g = (const float*)d_in[5];
    const float* ln2b = (const float*)d_in[6];
    const float* Wup  = (const float*)d_in[7];
    const float* bup  = (const float*)d_in[8];
    const float* Wh   = (const float*)d_in[9];
    const float* bh   = (const float*)d_in[10];
    const float* Wdn  = (const float*)d_in[11];
    const float* bdn  = (const float*)d_in[12];
    float* out = (float*)d_out;

    const long MS = (long)Bt * S;                // 8192 rows

    // bump allocator over workspace, 256B aligned; aliasing by lifetime
    char* ws = (char*)d_ws;
    auto alloc = [&](size_t bytes) -> char* {
        char* p = ws;
        ws += (bytes + 255) & ~(size_t)255;
        return p;
    };
    __bf16* h_bf   = (__bf16*)alloc(MS * Dm * 2);          // ln1(x); reused for ln2(x1)
    __bf16* q_bf   = (__bf16*)alloc(MS * Dm * 2);
    __bf16* vT_bf  = (__bf16*)alloc(MS * Dm * 2);          // V^T [B][Dm][S]
    char*   big0   = alloc((size_t)Bt * S * S * 4);        // scores f32 -> a2 bf16
    char*   big1   = alloc((size_t)MS * Dh * 2);           // probs bf16 -> a1 bf16
    float*  x1     = (float*)alloc(MS * Dm * 4);
    __bf16* qkT_bf = (__bf16*)alloc((size_t)Dm * Dm * 2);
    __bf16* ovT_bf = (__bf16*)alloc((size_t)Dm * Dm * 2);
    __bf16* Wup_bf = (__bf16*)alloc((size_t)Dh * Dm * 2);
    __bf16* Wh_bf  = (__bf16*)alloc((size_t)Dh * Dh * 2);
    __bf16* Wdn_bf = (__bf16*)alloc((size_t)Dm * Dh * 2);
    float*  scores = (float*)big0;
    __bf16* probs  = (__bf16*)big1;
    __bf16* a1     = (__bf16*)big1;
    __bf16* a2     = (__bf16*)big0;

    // ---- 0) weight conversion (bf16 W_h fits L2 -> GEMM B reads hit L2) ----
    cvt_bf16_kernel<<<4096, 256, 0, stream>>>(Wup, Wup_bf, (long)Dh * Dm);
    cvt_bf16_kernel<<<4096, 256, 0, stream>>>(Wh,  Wh_bf,  (long)Dh * Dh);
    cvt_bf16_kernel<<<4096, 256, 0, stream>>>(Wdn, Wdn_bf, (long)Dm * Dh);
    dim3 tb(32, 8), tg(Dm / 32, Dm / 32);
    cvtT_bf16_kernel<<<tg, tb, 0, stream>>>(qk, qkT_bf, Dm, Dm);
    cvtT_bf16_kernel<<<tg, tb, 0, stream>>>(ov, ovT_bf, Dm, Dm);

    // ---- 1) h = ln1(x) ----
    ln_bf16_kernel<<<(int)MS, 256, 0, stream>>>(x, ln1g, ln1b, h_bf);

    // ---- 2) q = h @ qk ; v^T = (h @ ov)^T ----
    dim3 gProj(Dm / BN, (int)(MS / BM), 1);
    gemm_bf16_wmma<OUT_BF16, false, false, false, false>
        <<<gProj, 256, 0, stream>>>(h_bf, qkT_bf, nullptr, nullptr, q_bf,
                                    (int)MS, Dm, Dm, 0, 0, 0, 0);
    gemm_bf16_wmma<OUT_BF16_T, false, false, false, false>
        <<<gProj, 256, 0, stream>>>(h_bf, ovT_bf, nullptr, nullptr, vT_bf,
                                    (int)MS, Dm, Dm, 0, 0, 0, 0);

    // ---- 3) scores = q @ h^T / sqrt(D), causal blocks skipped ----
    dim3 gScore(S / BN, S / BM, Bt);
    gemm_bf16_wmma<OUT_SCORES, false, false, true, false>
        <<<gScore, 256, 0, stream>>>(q_bf, h_bf, nullptr, nullptr, scores,
                                     S, S, Dm,
                                     (long)S * Dm, (long)S * Dm, (long)S * S, 0);

    // ---- 4) probs = causal softmax(scores) ----
    softmax_bf16_kernel<<<(int)(Bt * S), 256, 0, stream>>>(scores, probs);

    // ---- 5) x1 = x + probs @ V  (K clamped to causal extent) ----
    dim3 gAttn(Dm / BN, S / BM, Bt);
    gemm_bf16_wmma<OUT_F32_RESID, false, false, false, true>
        <<<gAttn, 256, 0, stream>>>(probs, vT_bf, nullptr, x, x1,
                                    S, Dm, S,
                                    (long)S * S, (long)Dm * S,
                                    (long)S * Dm, (long)S * Dm);

    // ---- 6) h2 = ln2(x1) ----
    ln_bf16_kernel<<<(int)MS, 256, 0, stream>>>(x1, ln2g, ln2b, h_bf);

    // ---- 7) MLP ----
    dim3 gUp(Dh / BN, (int)(MS / BM), 1);
    gemm_bf16_wmma<OUT_BF16, true, true, false, false>
        <<<gUp, 256, 0, stream>>>(h_bf, Wup_bf, bup, nullptr, a1,
                                  (int)MS, Dh, Dm, 0, 0, 0, 0);
    gemm_bf16_wmma<OUT_BF16, true, true, false, false>
        <<<gUp, 256, 0, stream>>>(a1, Wh_bf, bh, nullptr, a2,
                                  (int)MS, Dh, Dh, 0, 0, 0, 0);
    dim3 gDn(Dm / BN, (int)(MS / BM), 1);
    gemm_bf16_wmma<OUT_F32_RESID, true, true, false, false>
        <<<gDn, 256, 0, stream>>>(a2, Wdn_bf, bdn, x1, out,
                                  (int)MS, Dm, Dh, 0, 0, 0, 0);
    (void)in_sizes; (void)n_in; (void)out_size; (void)ws_size;
}
